// StreetViewEncoder_9672266351006
// MI455X (gfx1250) — compile-verified, matrix operand order checked by reading, not run on hardware
//
#include <hip/hip_runtime.h>
#include <hip/hip_bf16.h>
#include <math.h>

// ---------------- problem constants (from reference setup_inputs) ------------
#define VIEWS   20
#define IMH     128
#define IMW     256
#define IMC     128
#define FEATC   96
#define NSCALES 32
#define KSEL    4
#define DSTATS  385       // 4*96 + 1
#define DPAD    416       // 13 * 32  (K padded for WMMA)
#define HID1    256
#define HID2    128
#define KT1     13        // DPAD / 32
#define NT1     16        // HID1 / 16
#define KT2     8         // HID1 / 32
#define NT2     8         // HID2 / 16
#define TP      32        // points per workgroup (2 M-tiles of 16)

#define W1P_HALFS (KT1 * NT1 * 512)   // 106496 halves
#define W2P_HALFS (KT2 * NT2 * 512)   // 32768 halves

typedef _Float16 v16h __attribute__((ext_vector_type(16)));
typedef float    v8f  __attribute__((ext_vector_type(8)));

union AFrag { unsigned u[8]; v16h v; };

// k offset of the low half of 32-bit pair 'tt' (0..7) for a given lane,
// per the CDNA5 16-bit A/B 16x32 VGPR layout (lanes 0-15: K 0..7,16..23;
// lanes 16-31: K 8..15,24..31; two consecutive K per VGPR).
__device__ __forceinline__ int kmap_pair(int tt, int lane) {
    int base = (tt < 4) ? (2 * tt) : (16 + 2 * (tt - 4));
    if (lane & 16) base += 8;
    return base;
}

// ---------------- weight repack kernels (f32 -> f16 WMMA B fragments) --------
__global__ void pack_w1(const float* __restrict__ W1, _Float16* __restrict__ w1p) {
    int idx = blockIdx.x * blockDim.x + threadIdx.x;
    if (idx >= W1P_HALFS) return;
    int j    = idx & 15;
    int lane = (idx >> 4) & 31;
    int nt   = (idx >> 9) & 15;
    int kt   = idx >> 13;
    int kb   = kmap_pair(j >> 1, lane) + (j & 1);
    int K    = kt * 32 + kb;
    int Ncol = nt * 16 + (lane & 15);
    float v  = (K < DSTATS) ? W1[K * HID1 + Ncol] : 0.0f;
    w1p[idx] = (_Float16)v;
}

__global__ void pack_w2(const float* __restrict__ W2, _Float16* __restrict__ w2p) {
    int idx = blockIdx.x * blockDim.x + threadIdx.x;
    if (idx >= W2P_HALFS) return;
    int j    = idx & 15;
    int lane = (idx >> 4) & 31;
    int nt   = (idx >> 9) & 7;
    int kt   = idx >> 12;
    int kb   = kmap_pair(j >> 1, lane) + (j & 1);
    int K    = kt * 32 + kb;
    int Ncol = nt * 16 + (lane & 15);
    w2p[idx] = (_Float16)W2[K * HID2 + Ncol];
}

// ---------------- fused gather + stats + WMMA MLP kernel ---------------------
__global__ __launch_bounds__(256)
void sve_fused(const float* __restrict__ fimg, const float* __restrict__ xyz,
               const float* __restrict__ R,    const float* __restrict__ t,
               const float* __restrict__ cam,  const float* __restrict__ b1,
               const float* __restrict__ b2,
               const _Float16* __restrict__ w1p, const _Float16* __restrict__ w2p,
               float* __restrict__ out, int npts)
{
    __shared__ _Float16 s_stats[TP * DPAD];   // fp16 stats, A-matrix staging
    __shared__ _Float16 s_h[TP * HID1];       // fp16 hidden, A-matrix staging
    __shared__ float    s_sc[8][KSEL][NSCALES];
    __shared__ int      s_valid[TP];

    const int tid  = threadIdx.x;
    const int lane = tid & 31;
    const int wv   = tid >> 5;

    // ======================= Phase 1: per-point stats ========================
    for (int i = 0; i < TP / 8; ++i) {
        const int pl = wv * (TP / 8) + i;
        const int gp = blockIdx.x * TP + pl;
        const bool inb = gp < npts;
        const int gpc = inb ? gp : 0;

        const float px = xyz[gpc * 3 + 0];
        const float py = xyz[gpc * 3 + 1];
        const float pz = xyz[gpc * 3 + 2];

        // --- per-lane view evaluation: key = (dist_bits, lane) ---
        unsigned mykey;
        {
            const int v = lane;
            if (v < VIEWS) {
                const float dx = px - t[v * 3 + 0];
                const float dy = py - t[v * 3 + 1];
                const float dz = pz - t[v * 3 + 2];
                const float* Rv = R + v * 9;
                const float p0 = Rv[0] * dx + Rv[3] * dy + Rv[6] * dz;
                const float p1 = Rv[1] * dx + Rv[4] * dy + Rv[7] * dz;
                const float p2 = Rv[2] * dx + Rv[5] * dy + Rv[8] * dz;
                const float zs = (p2 > 1e-5f) ? p2 : 1.0f;
                const float uc = p0 / zs * cam[v * 4 + 0] + cam[v * 4 + 2];
                const float vr = p1 / zs * cam[v * 4 + 1] + cam[v * 4 + 3];
                const bool vis = (p2 > 1e-3f) && (uc >= 0.0f) && (uc <= (float)(IMW - 1))
                                              && (vr >= 0.0f) && (vr <= (float)(IMH - 1));
                const float dist = sqrtf(dx * dx + dy * dy + dz * dz);
                mykey = vis ? __float_as_uint(dist) : 0x7f800000u;  // inf if hidden
            } else {
                mykey = 0xFFFFFFFFu;  // lanes >= VIEWS never selected
            }
        }

        // --- wave-wide top-K (smallest dist, lower index tie-break) ---
        int selv[KSEL];
        unsigned keycur = mykey;
        #pragma unroll
        for (int k = 0; k < KSEL; ++k) {
            unsigned d = keycur;
            unsigned l = (unsigned)lane;
            #pragma unroll
            for (int off = 16; off > 0; off >>= 1) {
                unsigned od = __shfl_xor((int)d, off, 32);
                unsigned ol = __shfl_xor((int)l, off, 32);
                if (od < d || (od == d && ol < l)) { d = od; l = ol; }
            }
            selv[k] = (int)l;
            if (lane == (int)l) keycur = 0xFFFFFFFFu;
        }

        // --- bilinear gather: 128 channels across 32 lanes (float4/lane) ---
        float fp[KSEL][4];
        bool  visk[KSEL];
        float depthk[KSEL];
        #pragma unroll
        for (int k = 0; k < KSEL; ++k) {
            const int v = selv[k];
            const float dx = px - t[v * 3 + 0];
            const float dy = py - t[v * 3 + 1];
            const float dz = pz - t[v * 3 + 2];
            const float* Rv = R + v * 9;
            const float p0 = Rv[0] * dx + Rv[3] * dy + Rv[6] * dz;
            const float p1 = Rv[1] * dx + Rv[4] * dy + Rv[7] * dz;
            const float p2 = Rv[2] * dx + Rv[5] * dy + Rv[8] * dz;
            const float zs = (p2 > 1e-5f) ? p2 : 1.0f;
            const float uc = p0 / zs * cam[v * 4 + 0] + cam[v * 4 + 2];
            const float vr = p1 / zs * cam[v * 4 + 1] + cam[v * 4 + 3];
            visk[k]   = (p2 > 1e-3f) && (uc >= 0.0f) && (uc <= (float)(IMW - 1))
                                     && (vr >= 0.0f) && (vr <= (float)(IMH - 1));
            depthk[k] = p2;

            float pr = fmaxf(fminf(vr - 0.5f, (float)(IMH - 1)), 0.0f);
            float pc = fmaxf(fminf(uc - 0.5f, (float)(IMW - 1)), 0.0f);
            const int r0 = (int)floorf(pr), c0 = (int)floorf(pc);
            const int r1 = (r0 + 1 < IMH - 1) ? r0 + 1 : IMH - 1;
            const int c1 = (c0 + 1 < IMW - 1) ? c0 + 1 : IMW - 1;
            const float wr1 = pr - (float)r0, wr0 = 1.0f - wr1;
            const float wc1 = pc - (float)c0, wc0 = 1.0f - wc1;

            const float* vbase = fimg + (size_t)v * IMH * IMW * IMC;
            float a0 = 0.f, a1 = 0.f, a2 = 0.f, a3 = 0.f;
            if (inb) {
                #pragma unroll
                for (int tap = 0; tap < 4; ++tap) {
                    const int rr = (tap & 2) ? r1 : r0;
                    const int cc = (tap & 1) ? c1 : c0;
                    const float w = ((tap & 2) ? wr1 : wr0) * ((tap & 1) ? wc1 : wc0);
                    const float4 val = *((const float4*)(vbase + ((size_t)rr * IMW + cc) * IMC) + lane);
                    a0 += w * val.x; a1 += w * val.y; a2 += w * val.z; a3 += w * val.w;
                }
            }
            fp[k][0] = a0; fp[k][1] = a1; fp[k][2] = a2; fp[k][3] = a3;
            if (lane >= 24) {   // scale channels 96..127 live in lanes 24..31
                const int s = 4 * (lane - 24);
                s_sc[wv][k][s + 0] = a0; s_sc[wv][k][s + 1] = a1;
                s_sc[wv][k][s + 2] = a2; s_sc[wv][k][s + 3] = a3;
            }
        }
        __syncthreads();   // publish s_sc (uniform trip count -> legal)

        // --- depth-binned scores + masked softmax ---
        const float inv_log64 = 1.0f / logf(64.0f);
        float score[KSEL];
        #pragma unroll
        for (int k = 0; k < KSEL; ++k) {
            const float d  = fminf(fmaxf(depthk[k], 1.0f), 64.0f);
            float pos = logf(d) * inv_log64 * (float)(NSCALES - 1);
            pos = fminf(fmaxf(pos, 0.0f), (float)(NSCALES - 1));
            const int  b0 = (int)floorf(pos);
            const int  bh = (b0 + 1 < NSCALES - 1) ? b0 + 1 : NSCALES - 1;
            const float fr = pos - (float)b0;
            score[k] = (1.0f - fr) * s_sc[wv][k][b0] + fr * s_sc[wv][k][bh];
        }
        const bool vany = visk[0] | visk[1] | visk[2] | visk[3];
        bool valk[KSEL];
        #pragma unroll
        for (int k = 0; k < KSEL; ++k) valk[k] = vany ? visk[k] : true;

        float mreduce = -INFINITY;
        #pragma unroll
        for (int k = 0; k < KSEL; ++k) if (valk[k]) mreduce = fmaxf(mreduce, score[k]);
        float e[KSEL], se = 0.0f;
        #pragma unroll
        for (int k = 0; k < KSEL; ++k) { e[k] = valk[k] ? expf(score[k] - mreduce) : 0.0f; se += e[k]; }
        const float inv_se = 1.0f / se;

        // --- stats into LDS (f16, row stride DPAD) ---
        if (lane < 24) {
            #pragma unroll
            for (int c = 0; c < 4; ++c) {
                float mean = 0.0f;
                #pragma unroll
                for (int k = 0; k < KSEL; ++k) mean += e[k] * inv_se * fp[k][c];
                float var = 0.0f;
                #pragma unroll
                for (int k = 0; k < KSEL; ++k) {
                    const float dd = fp[k][c] - mean;
                    var += e[k] * inv_se * dd * dd;
                }
                float mx = -INFINITY, mn = INFINITY;
                #pragma unroll
                for (int k = 0; k < KSEL; ++k) if (valk[k]) {
                    mx = fmaxf(mx, fp[k][c]); mn = fminf(mn, fp[k][c]);
                }
                const int ch = 4 * lane + c;
                s_stats[pl * DPAD +            ch] = (_Float16)mean;
                s_stats[pl * DPAD + FEATC    + ch] = (_Float16)var;
                s_stats[pl * DPAD + 2*FEATC  + ch] = (_Float16)mx;
                s_stats[pl * DPAD + 3*FEATC  + ch] = (_Float16)mn;
            }
        } else {
            if (lane == 24) {
                float smax = -INFINITY;
                #pragma unroll
                for (int k = 0; k < KSEL; ++k) if (valk[k]) smax = fmaxf(smax, score[k]);
                s_stats[pl * DPAD + 4*FEATC] = (_Float16)smax;
            }
            for (int q = DSTATS + (lane - 24); q < DPAD; q += 8)
                s_stats[pl * DPAD + q] = (_Float16)0.0f;
        }
        if (lane == 0) s_valid[pl] = (inb && vany) ? 1 : 0;
    }
    __syncthreads();

    // ===================== Phase 2: MLP layer 1 (385->256) ===================
    // 2 M-tiles x 16 N-tiles = 32 tile pairs over 8 waves.
    for (int pair = wv; pair < 2 * NT1; pair += 8) {
        const int mt = pair & 1;
        const int nt = pair >> 1;
        v8f acc = {0.f, 0.f, 0.f, 0.f, 0.f, 0.f, 0.f, 0.f};
        const int row  = mt * 16 + (lane & 15);
        #pragma unroll
        for (int kt = 0; kt < KT1; ++kt) {
            AFrag af;
            #pragma unroll
            for (int tt = 0; tt < 8; ++tt) {
                const int kb = kmap_pair(tt, lane);
                af.u[tt] = *(const unsigned*)&s_stats[row * DPAD + kt * 32 + kb];
            }
            const v16h bfrag = *(const v16h*)&w1p[((kt * NT1 + nt) * 32 + lane) * 16];
            acc = __builtin_amdgcn_wmma_f32_16x16x32_f16(
                      false, af.v, false, bfrag, (short)0, acc, false, false);
        }
        const float bias  = b1[nt * 16 + (lane & 15)];
        const int   mbase = (lane & 16) ? 8 : 0;
        const int   ncol  = lane & 15;
        #pragma unroll
        for (int r = 0; r < 8; ++r) {
            float v = acc[r] + bias;
            v = (v > 0.0f) ? v : 0.0f;   // ReLU
            s_h[(mt * 16 + mbase + r) * HID1 + nt * 16 + ncol] = (_Float16)v;
        }
    }
    __syncthreads();

    // ===================== Phase 3: MLP layer 2 (256->128) ===================
    for (int pair = wv; pair < 2 * NT2; pair += 8) {
        const int mt = pair & 1;
        const int nt = pair >> 1;
        v8f acc = {0.f, 0.f, 0.f, 0.f, 0.f, 0.f, 0.f, 0.f};
        const int row = mt * 16 + (lane & 15);
        #pragma unroll
        for (int kt = 0; kt < KT2; ++kt) {
            AFrag af;
            #pragma unroll
            for (int tt = 0; tt < 8; ++tt) {
                const int kb = kmap_pair(tt, lane);
                af.u[tt] = *(const unsigned*)&s_h[row * HID1 + kt * 32 + kb];
            }
            const v16h bfrag = *(const v16h*)&w2p[((kt * NT2 + nt) * 32 + lane) * 16];
            acc = __builtin_amdgcn_wmma_f32_16x16x32_f16(
                      false, af.v, false, bfrag, (short)0, acc, false, false);
        }
        const float bias  = b2[nt * 16 + (lane & 15)];
        const int   mbase = (lane & 16) ? 8 : 0;
        const int   ncol  = lane & 15;
        #pragma unroll
        for (int r = 0; r < 8; ++r) {
            const int pl = mt * 16 + mbase + r;
            const int gp = blockIdx.x * TP + pl;
            if (gp < npts) {
                const float v = s_valid[pl] ? (acc[r] + bias) : 0.0f;
                out[(size_t)gp * HID2 + nt * 16 + ncol] = v;
            }
        }
    }
}

// ---------------- host-side launcher ----------------------------------------
extern "C" void kernel_launch(void* const* d_in, const int* in_sizes, int n_in,
                              void* d_out, int out_size, void* d_ws, size_t ws_size,
                              hipStream_t stream) {
    const float* fimg = (const float*)d_in[0];
    const float* xyz  = (const float*)d_in[1];
    const float* R    = (const float*)d_in[2];
    const float* t    = (const float*)d_in[3];
    const float* cam  = (const float*)d_in[4];
    const float* W1   = (const float*)d_in[5];
    const float* b1   = (const float*)d_in[6];
    const float* W2   = (const float*)d_in[7];
    const float* b2   = (const float*)d_in[8];
    float* out = (float*)d_out;

    const int npts = in_sizes[1] / 3;

    // workspace: packed f16 weights (W1 ~208KB, W2 ~64KB)
    _Float16* w1p = (_Float16*)d_ws;
    _Float16* w2p = w1p + W1P_HALFS;

    pack_w1<<<(W1P_HALFS + 255) / 256, 256, 0, stream>>>(W1, w1p);
    pack_w2<<<(W2P_HALFS + 255) / 256, 256, 0, stream>>>(W2, w2p);

    const int grid = (npts + TP - 1) / TP;
    sve_fused<<<grid, 256, 0, stream>>>(fimg, xyz, R, t, cam, b1, b2,
                                        w1p, w2p, out, npts);
}
